// CFGCN_30614526886035
// MI455X (gfx1250) — matrix operand order repaired and checked by reading, not compiled, not run on hardware
//
#include <hip/hip_runtime.h>

// ---------------------------------------------------------------------------
// CDNA5 (gfx1250, wave32) CFGCN forward pass.
// - All 1x1 convs / graph matmuls -> v_wmma_f32_16x16x32_bf16, fused
//   bias / folded-BN / residual / ReLU epilogues.
// - Conv-style GEMMs stage their B tile (32K x 64N) into LDS with the
//   Tensor Data Mover (tensor_load_to_lds + s_wait_tensorcnt), wave 0 issues.
// - A fragments use global_load_b128 fast path when contiguous.
// - Concat operand of the final GEMM kept in bf16 (memory-bound graph).
// ---------------------------------------------------------------------------

typedef __attribute__((ext_vector_type(16))) __bf16        v16bf;
typedef __attribute__((ext_vector_type(8)))  float         v8f;
typedef __attribute__((ext_vector_type(4)))  unsigned int  v4u;
typedef __attribute__((ext_vector_type(8)))  int           v8i;
typedef __attribute__((ext_vector_type(4)))  int           v4i;

#define BSZ   4
#define PL    256      // planes
#define IP    128      // inter planes
#define RR    64       // planes / ratio
#define HH    128
#define WW    128
#define NPIX  (HH * WW)   // 16384
#define NL    256         // 16*16 pixels of the local branch

// ---------------------------------------------------------------------------
// Fold BN(g,b,m,v) -> scale/shift:  y = x*scale + shift
// ---------------------------------------------------------------------------
__global__ void bn_fold_k(const float* __restrict__ g, const float* __restrict__ b,
                          const float* __restrict__ m, const float* __restrict__ v,
                          float* __restrict__ sc, float* __restrict__ sh, int n) {
    int i = blockIdx.x * blockDim.x + threadIdx.x;
    if (i < n) {
        float s = g[i] * rsqrtf(v[i] + 1e-5f);
        sc[i] = s;
        sh[i] = b[i] - m[i] * s;
    }
}

// ---------------------------------------------------------------------------
// Depthwise 3x3 stride-2 pad-1 conv + folded BN
// ---------------------------------------------------------------------------
__global__ void dwconv_bn_k(const float* __restrict__ x, const float* __restrict__ w,
                            const float* __restrict__ sc, const float* __restrict__ sh,
                            float* __restrict__ y, int Bn, int C, int Hi, int Wi,
                            int Ho, int Wo) {
    long total = (long)Bn * C * Ho * Wo;
    long i = (long)blockIdx.x * blockDim.x + threadIdx.x;
    if (i >= total) return;
    int ox = (int)(i % Wo); long t = i / Wo;
    int oy = (int)(t % Ho); t /= Ho;
    int c  = (int)(t % C);  int b = (int)(t / C);
    const float* xp = x + ((long)b * C + c) * Hi * Wi;
    const float* wp = w + (long)c * 9;
    float s = 0.f;
#pragma unroll
    for (int ky = 0; ky < 3; ++ky) {
        int iy = oy * 2 - 1 + ky;
        if (iy < 0 || iy >= Hi) continue;
#pragma unroll
        for (int kx = 0; kx < 3; ++kx) {
            int ix = ox * 2 - 1 + kx;
            if (ix < 0 || ix >= Wi) continue;
            s += xp[(long)iy * Wi + ix] * wp[ky * 3 + kx];
        }
    }
    y[i] = s * sc[c] + sh[c];
}

// ---------------------------------------------------------------------------
// A fragment (16x32 bf16, ISA 7.12.2 layout). Fast b128 path when aCS == 1.
// row = global M row for this lane; slot e <-> K = k0+(e&7)+((e&8)<<1)+8*half
// ---------------------------------------------------------------------------
__device__ __forceinline__ v16bf load_afrag(const float* __restrict__ Ab, int row,
                                            int aRS, int aCS, int k0, int half) {
    v16bf af;
    if (aCS == 1) {
        const float* p = Ab + (long)row * aRS + k0 + 8 * half;
        float4 a0 = ((const float4*)p)[0];
        float4 a1 = ((const float4*)p)[1];
        float4 a2 = ((const float4*)(p + 16))[0];
        float4 a3 = ((const float4*)(p + 16))[1];
        af[0] = (__bf16)a0.x; af[1] = (__bf16)a0.y; af[2]  = (__bf16)a0.z; af[3]  = (__bf16)a0.w;
        af[4] = (__bf16)a1.x; af[5] = (__bf16)a1.y; af[6]  = (__bf16)a1.z; af[7]  = (__bf16)a1.w;
        af[8] = (__bf16)a2.x; af[9] = (__bf16)a2.y; af[10] = (__bf16)a2.z; af[11] = (__bf16)a2.w;
        af[12] = (__bf16)a3.x; af[13] = (__bf16)a3.y; af[14] = (__bf16)a3.z; af[15] = (__bf16)a3.w;
    } else {
#pragma unroll
        for (int e = 0; e < 16; ++e) {
            int kk = k0 + (e & 7) + ((e & 8) << 1) + 8 * half;
            af[e] = (__bf16)Ab[(long)row * aRS + (long)kk * aCS];
        }
    }
    return af;
}

__device__ __forceinline__ unsigned rfl(unsigned x) {
    return (unsigned)__builtin_amdgcn_readfirstlane((int)x);
}

// ---------------------------------------------------------------------------
// WMMA GEMM:  Y[b, m, n] = sum_k A[b](m,k) * B[b](k,n)     (bf16 x bf16 -> f32)
// A element (m,k) = A[b*aB + m*aRS + k*aCS]  (f32, cvt to bf16)
// B element (k,n) = B[b*bB + k*bRS + n*bCS]  (f32 or bf16)
// tdm != 0 (requires bCS==1): B tile (32K x 64N) staged to LDS by the Tensor
// Data Mover; otherwise direct per-lane gathers.
// Epilogue: (+bias[m]) -> (*bnSc[m] + bnSh[m]) -> (+resid[b,m,n]) -> relu
// One wave = 64x16 macro-tile. M%64==0, N%64==0, K%32==0 at all call sites.
// ---------------------------------------------------------------------------
template <typename TB, typename TOUT>
__global__ __launch_bounds__(128)
void wmma_gemm_k(const float* __restrict__ A, long aB, int aRS, int aCS,
                 const TB* __restrict__ Bm, long bB, int bRS, int bCS,
                 TOUT* __restrict__ Y, long yB, int yRS,
                 const float* __restrict__ bias,
                 const float* __restrict__ bnSc, const float* __restrict__ bnSh,
                 const float* __restrict__ resid, long rB, int rRS,
                 int M, int N, int K, int relu, int tdm) {
    __shared__ __align__(16) char smRaw[32 * 64 * sizeof(float)];
    TB* smB = (TB*)smRaw;

    const int lane = threadIdx.x & 31;
    const int wave = threadIdx.x >> 5;
    const int bat  = blockIdx.z;
    const int nblk = blockIdx.x * 64;
    const int n0   = nblk + wave * 16;
    const int m0   = blockIdx.y * 64;
    const int half = lane >> 4;      // 0: lanes 0-15, 1: lanes 16-31
    const int ln   = lane & 15;

    const float* Ab = A  + (long)bat * aB;
    const TB*    Bb = Bm + (long)bat * bB;

    v8f acc[4];
#pragma unroll
    for (int t = 0; t < 4; ++t)
#pragma unroll
        for (int r = 0; r < 8; ++r) acc[t][r] = 0.f;

    if (tdm) {
        // ---- Tensor Data Mover staged B operand ----
        const unsigned dsz   = (sizeof(TB) == 4) ? 2u : 1u;   // D# data_size code
        const unsigned ldsa  = rfl((unsigned)(unsigned long long)(uintptr_t)smB);
        const unsigned td0   = rfl((unsigned)bRS);            // tensor_dim0
        const unsigned td1   = rfl((unsigned)K);              // tensor_dim1
        const unsigned long long str0 = (unsigned long long)(unsigned)bRS;

        for (int k0 = 0; k0 < K; k0 += 32) {
            if (wave == 0) {
                unsigned long long ga =
                    (unsigned long long)(uintptr_t)(const void*)(Bb + (long)k0 * bRS + nblk);
                v4u g0;
                g0.x = rfl(1u);                                  // count=1, user mode
                g0.y = ldsa;                                     // lds_addr
                g0.z = rfl((unsigned)(ga & 0xFFFFFFFFu));        // global_addr[31:0]
                g0.w = rfl((unsigned)((ga >> 32) & 0x1FFFFFFu) | (2u << 30)); // [56:32]|type=2
                v8i g1 = {
                    (int)rfl(dsz << 16),                          // mask=0, data_size
                    (int)rfl((td0 & 0xFFFFu) << 16),              // tensor_dim0 lo
                    (int)rfl((td0 >> 16) | ((td1 & 0xFFFFu) << 16)),
                    (int)rfl((td1 >> 16) | (64u << 16)),          // tile_dim0 = 64
                    (int)rfl(32u),                                // tile_dim1 = 32, tile_dim2 = 0
                    (int)rfl((unsigned)(str0 & 0xFFFFFFFFu)),     // tensor_dim0_stride lo
                    (int)rfl((unsigned)(str0 >> 32)),             // stride hi, dim1_stride lo = 0
                    0
                };
                v4i z4 = {0, 0, 0, 0};
#if __has_include(<hip/amd_detail/amd_gfx1250_TDM.h>)
                v8i z8 = {0, 0, 0, 0, 0, 0, 0, 0};
                __builtin_amdgcn_tensor_load_to_lds(g0, g1, z4, z4, z8, 0);
#else
                __builtin_amdgcn_tensor_load_to_lds(g0, g1, z4, z4, 0);
#endif
                __builtin_amdgcn_s_wait_tensorcnt(0);
            }
            __syncthreads();
            v16bf bf;
#pragma unroll
            for (int e = 0; e < 16; ++e)
                bf[e] = (__bf16)(float)smB[((e & 7) + ((e & 8) << 1) + 8 * half) * 64 + (wave * 16 + ln)];
#pragma unroll
            for (int t = 0; t < 4; ++t) {
                v16bf af = load_afrag(Ab, m0 + t * 16 + ln, aRS, aCS, k0, half);
                acc[t] = __builtin_amdgcn_wmma_f32_16x16x32_bf16(
                    false, af, false, bf, (short)0, acc[t], false, false);
            }
            __syncthreads();
        }
    } else {
        // ---- direct per-lane gather path (used by the NT GEMM) ----
        const TB* bcol = Bb + (long)(n0 + ln) * bCS;
        for (int k0 = 0; k0 < K; k0 += 32) {
            v16bf bf;
#pragma unroll
            for (int e = 0; e < 16; ++e) {
                int kk = k0 + (e & 7) + ((e & 8) << 1) + 8 * half;
                bf[e] = (__bf16)(float)bcol[(long)kk * bRS];
            }
#pragma unroll
            for (int t = 0; t < 4; ++t) {
                v16bf af = load_afrag(Ab, m0 + t * 16 + ln, aRS, aCS, k0, half);
                acc[t] = __builtin_amdgcn_wmma_f32_16x16x32_bf16(
                    false, af, false, bf, (short)0, acc[t], false, false);
            }
        }
    }

    // ---- epilogue + store: D element (M = r + 8*half, N = ln) in vgpr r ----
#pragma unroll
    for (int t = 0; t < 4; ++t) {
#pragma unroll
        for (int r = 0; r < 8; ++r) {
            int mm = m0 + t * 16 + r + 8 * half;
            float val = acc[t][r];
            if (bias)  val += bias[mm];
            if (bnSc)  val = val * bnSc[mm] + bnSh[mm];
            if (resid) val += resid[(long)bat * rB + (long)mm * rRS + (n0 + ln)];
            if (relu)  val = fmaxf(val, 0.f);
            Y[(long)bat * yB + (long)mm * yRS + (n0 + ln)] = (TOUT)val;
        }
    }
}

// ---------------------------------------------------------------------------
// SpatialGCN attention: S[b,c,d] = sum_n q[b,c,n]*v[b,d,n]; A = softmax_d(S)
// ---------------------------------------------------------------------------
__global__ void qv_softmax_k(const float* __restrict__ q, const float* __restrict__ v,
                             float* __restrict__ Aatt) {
    const int c = blockIdx.x, b = blockIdx.y, d = threadIdx.x;
    const float* qc = q + ((long)b * IP + c) * NL;
    const float* vd = v + ((long)b * IP + d) * NL;
    float s = 0.f;
    for (int n = 0; n < NL; ++n) s += qc[n] * vd[n];
    __shared__ float red[IP];
    red[d] = s; __syncthreads();
    for (int st = IP / 2; st > 0; st >>= 1) {
        if (d < st) red[d] = fmaxf(red[d], red[d + st]);
        __syncthreads();
    }
    float mx = red[0]; __syncthreads();
    float e = expf(s - mx);
    red[d] = e; __syncthreads();
    for (int st = IP / 2; st > 0; st >>= 1) {
        if (d < st) red[d] += red[d + st];
        __syncthreads();
    }
    Aatt[((long)b * IP + c) * IP + d] = e / red[0];
}

// z = z_idt + bn_adj( adj_w @ z_idt^T )^T        ([B,128,64], adj 64x64)
__global__ void adj_z_k(const float* __restrict__ zi, const float* __restrict__ adj,
                        const float* __restrict__ sc, const float* __restrict__ sh,
                        float* __restrict__ z) {
    int i = blockIdx.x * blockDim.x + threadIdx.x;
    if (i >= BSZ * IP * RR) return;
    int d = i & (RR - 1);
    int c = (i >> 6) & (IP - 1);
    int b = i >> 13;
    const float* zr = zi + ((long)b * IP + c) * RR;
    float s = 0.f;
    for (int e = 0; e < RR; ++e) s += adj[d * RR + e] * zr[e];
    z[i] = zr[d] + s * sc[d] + sh[d];
}

// zf = bn_wg( wg_w @ z )                          ([B,128,64], wg 128x128)
__global__ void wg_z_k(const float* __restrict__ z, const float* __restrict__ wg,
                       const float* __restrict__ sc, const float* __restrict__ sh,
                       float* __restrict__ zf) {
    int i = blockIdx.x * blockDim.x + threadIdx.x;
    if (i >= BSZ * IP * RR) return;
    int d = i & (RR - 1);
    int o = (i >> 6) & (IP - 1);
    int b = i >> 13;
    float s = 0.f;
    for (int c = 0; c < IP; ++c) s += wg[o * IP + c] * z[((long)b * IP + c) * RR + d];
    zf[i] = s * sc[o] + sh[o];
}

// ---------------------------------------------------------------------------
// Bilinear upsample (align_corners) of local [B,256,16,16] -> gate with feat,
// write slf = feat*(local+1) into channel-half 0 of the bf16 concat buffer.
// ---------------------------------------------------------------------------
__global__ void upsample_gate_k(const float* __restrict__ feat, const float* __restrict__ loc,
                                __bf16* __restrict__ outc) {
    long total = (long)BSZ * PL * HH * WW;
    long i = (long)blockIdx.x * blockDim.x + threadIdx.x;
    if (i >= total) return;
    int ox = (int)(i % WW); long t = i / WW;
    int oy = (int)(t % HH); t /= HH;
    int c  = (int)(t % PL); int b = (int)(t / PL);
    const float sy = 15.f / 127.f;
    float fy = oy * sy, fx = ox * sy;
    int y0 = (int)floorf(fy), x0 = (int)floorf(fx);
    int y1 = min(y0 + 1, 15), x1 = min(x0 + 1, 15);
    float wy = fy - (float)y0, wx = fx - (float)x0;
    const float* lp = loc + ((long)b * PL + c) * 256;
    float v00 = lp[y0 * 16 + x0], v01 = lp[y0 * 16 + x1];
    float v10 = lp[y1 * 16 + x0], v11 = lp[y1 * 16 + x1];
    float lv = v00 * (1.f - wy) * (1.f - wx) + v01 * (1.f - wy) * wx +
               v10 * wy * (1.f - wx) + v11 * wy * wx;
    float f = feat[((long)b * PL + c) * NPIX + (long)oy * WW + ox];
    outc[(long)b * (2L * PL * NPIX) + (long)c * NPIX + (long)oy * WW + ox] =
        (__bf16)(f * lv + f);
}

// ---------------------------------------------------------------------------
// Host-side orchestration
// ---------------------------------------------------------------------------
extern "C" void kernel_launch(void* const* d_in, const int* in_sizes, int n_in,
                              void* d_out, int out_size, void* d_ws, size_t ws_size,
                              hipStream_t stream) {
    (void)in_sizes; (void)n_in; (void)out_size; (void)ws_size;
    const float* feat     = (const float*)d_in[0];
    const float* lc_w[3]  = {(const float*)d_in[1], (const float*)d_in[2], (const float*)d_in[3]};
    const int    bn_lc[3] = {4, 8, 12};
    const float* sg_k_w = (const float*)d_in[16]; const float* sg_k_b = (const float*)d_in[17];
    const float* sg_v_w = (const float*)d_in[18]; const float* sg_v_b = (const float*)d_in[19];
    const float* sg_q_w = (const float*)d_in[20]; const float* sg_q_b = (const float*)d_in[21];
    const float* sg_wg_w  = (const float*)d_in[22];   const int bn_sgwg  = 23;
    const float* sg_out_w = (const float*)d_in[27];
    const float* sg_out_b = (const float*)d_in[28];   const int bn_sgout = 29;
    const float* phi_w    = (const float*)d_in[33];   const int bn_phi   = 34;
    const float* theta_w  = (const float*)d_in[38];   const int bn_theta = 39;
    const float* adj_w    = (const float*)d_in[43];   const int bn_adj   = 44;
    const float* wg_w     = (const float*)d_in[48];   const int bn_wg    = 49;
    const float* conv3_w  = (const float*)d_in[53];   const int bn3      = 54;
    const float* final_w  = (const float*)d_in[58];   const int bn_final = 59;
    float* outp = (float*)d_out;

    // -------- workspace carving (deterministic) --------
    char* ws = (char*)d_ws;
    size_t off = 0;
    auto allocf = [&](size_t n) -> float* {
        off = (off + 255) & ~(size_t)255;
        float* p = (float*)(ws + off); off += n * sizeof(float); return p;
    };
    auto allocb = [&](size_t n) -> __bf16* {
        off = (off + 255) & ~(size_t)255;
        __bf16* p = (__bf16*)(ws + off); off += n * sizeof(__bf16); return p;
    };

    float *sc_lc[3], *sh_lc[3];
    for (int i = 0; i < 3; ++i) { sc_lc[i] = allocf(PL); sh_lc[i] = allocf(PL); }
    float *sc_sgwg = allocf(IP), *sh_sgwg = allocf(IP);
    float *sc_sgo  = allocf(PL), *sh_sgo  = allocf(PL);
    float *sc_phi  = allocf(IP), *sh_phi  = allocf(IP);
    float *sc_th   = allocf(RR), *sh_th   = allocf(RR);
    float *sc_adj  = allocf(RR), *sh_adj  = allocf(RR);
    float *sc_wg   = allocf(IP), *sh_wg   = allocf(IP);
    float *sc_3    = allocf(PL), *sh_3    = allocf(PL);
    float *sc_fin  = allocf(PL), *sh_fin  = allocf(PL);

    float* l0  = allocf((size_t)BSZ * PL * 64 * 64);
    float* l1  = allocf((size_t)BSZ * PL * 32 * 32);
    float* lf  = allocf((size_t)BSZ * PL * NL);
    float* nk  = allocf((size_t)BSZ * IP * NL);
    float* nv  = allocf((size_t)BSZ * IP * NL);
    float* nq  = allocf((size_t)BSZ * IP * NL);
    float* Aat = allocf((size_t)BSZ * IP * IP);
    float* AV  = allocf((size_t)BSZ * IP * NL);
    float* AVW = allocf((size_t)BSZ * IP * NL);
    float* lsg = allocf((size_t)BSZ * PL * NL);
    float* xsq = allocf((size_t)BSZ * IP * NPIX);
    float* bmb = allocf((size_t)BSZ * RR * NPIX);
    float* zid = allocf((size_t)BSZ * IP * RR);
    float* zz  = allocf((size_t)BSZ * IP * RR);
    float* zf  = allocf((size_t)BSZ * IP * RR);
    float* ybf = allocf((size_t)BSZ * IP * NPIX);
    __bf16* cat = allocb((size_t)BSZ * 2 * PL * NPIX);

    // -------- fold all BN params --------
    auto fold = [&](int gi, float* sc, float* sh, int n) {
        bn_fold_k<<<1, 256, 0, stream>>>((const float*)d_in[gi], (const float*)d_in[gi + 1],
                                         (const float*)d_in[gi + 2], (const float*)d_in[gi + 3],
                                         sc, sh, n);
    };
    for (int i = 0; i < 3; ++i) fold(bn_lc[i], sc_lc[i], sh_lc[i], PL);
    fold(bn_sgwg, sc_sgwg, sh_sgwg, IP);  fold(bn_sgout, sc_sgo, sh_sgo, PL);
    fold(bn_phi, sc_phi, sh_phi, IP);     fold(bn_theta, sc_th, sh_th, RR);
    fold(bn_adj, sc_adj, sh_adj, RR);     fold(bn_wg, sc_wg, sh_wg, IP);
    fold(bn3, sc_3, sh_3, PL);            fold(bn_final, sc_fin, sh_fin, PL);

    // -------- local branch: 3x depthwise stride-2 + BN --------
    {
        long t0 = (long)BSZ * PL * 64 * 64;
        dwconv_bn_k<<<dim3((t0 + 255) / 256), 256, 0, stream>>>(
            feat, lc_w[0], sc_lc[0], sh_lc[0], l0, BSZ, PL, HH, WW, 64, 64);
        long t1 = (long)BSZ * PL * 32 * 32;
        dwconv_bn_k<<<dim3((t1 + 255) / 256), 256, 0, stream>>>(
            l0, lc_w[1], sc_lc[1], sh_lc[1], l1, BSZ, PL, 64, 64, 32, 32);
        long t2 = (long)BSZ * PL * 16 * 16;
        dwconv_bn_k<<<dim3((t2 + 255) / 256), 256, 0, stream>>>(
            l1, lc_w[2], sc_lc[2], sh_lc[2], lf, BSZ, PL, 32, 32, 16, 16);
    }

    const dim3 blk(128);
    auto grid = [](int N, int M) { return dim3(N / 64, M / 64, BSZ); };

    // -------- SpatialGCN on lf ([B,256,256]) --------
    wmma_gemm_k<float, float><<<grid(NL, IP), blk, 0, stream>>>(
        sg_k_w, 0, PL, 1, lf, (long)PL * NL, NL, 1, nk, (long)IP * NL, NL,
        sg_k_b, nullptr, nullptr, nullptr, 0, 0, IP, NL, PL, 0, 1);
    wmma_gemm_k<float, float><<<grid(NL, IP), blk, 0, stream>>>(
        sg_v_w, 0, PL, 1, lf, (long)PL * NL, NL, 1, nv, (long)IP * NL, NL,
        sg_v_b, nullptr, nullptr, nullptr, 0, 0, IP, NL, PL, 0, 1);
    wmma_gemm_k<float, float><<<grid(NL, IP), blk, 0, stream>>>(
        sg_q_w, 0, PL, 1, lf, (long)PL * NL, NL, 1, nq, (long)IP * NL, NL,
        sg_q_b, nullptr, nullptr, nullptr, 0, 0, IP, NL, PL, 0, 1);
    qv_softmax_k<<<dim3(IP, BSZ), IP, 0, stream>>>(nq, nv, Aat);
    // AV[b,d,n] = sum_c k[b,c,n] * Aat[b,c,d]   (A operand = Aat transposed)
    wmma_gemm_k<float, float><<<grid(NL, IP), blk, 0, stream>>>(
        Aat, (long)IP * IP, 1, IP, nk, (long)IP * NL, NL, 1, AV, (long)IP * NL, NL,
        nullptr, nullptr, nullptr, nullptr, 0, 0, IP, NL, IP, 0, 1);
    wmma_gemm_k<float, float><<<grid(NL, IP), blk, 0, stream>>>(
        sg_wg_w, 0, IP, 1, AV, (long)IP * NL, NL, 1, AVW, (long)IP * NL, NL,
        nullptr, sc_sgwg, sh_sgwg, nullptr, 0, 0, IP, NL, IP, 0, 1);
    // local = relu(bn(out_w @ AVW + out_b) + lf)
    wmma_gemm_k<float, float><<<grid(NL, PL), blk, 0, stream>>>(
        sg_out_w, 0, IP, 1, AVW, (long)IP * NL, NL, 1, lsg, (long)PL * NL, NL,
        sg_out_b, sc_sgo, sh_sgo, lf, (long)PL * NL, NL, PL, NL, IP, 1, 1);

    // slf = feat*(up(local)+1) -> concat channels [0,256)
    {
        long tt = (long)BSZ * PL * NPIX;
        upsample_gate_k<<<dim3((tt + 255) / 256), 256, 0, stream>>>(feat, lsg, cat);
    }

    // -------- coordinate GCN branch --------
    wmma_gemm_k<float, float><<<grid(NPIX, IP), blk, 0, stream>>>(
        phi_w, 0, PL, 1, feat, (long)PL * NPIX, NPIX, 1, xsq, (long)IP * NPIX, NPIX,
        nullptr, sc_phi, sh_phi, nullptr, 0, 0, IP, NPIX, PL, 0, 1);
    wmma_gemm_k<float, float><<<grid(NPIX, RR), blk, 0, stream>>>(
        theta_w, 0, PL, 1, feat, (long)PL * NPIX, NPIX, 1, bmb, (long)RR * NPIX, NPIX,
        nullptr, sc_th, sh_th, nullptr, 0, 0, RR, NPIX, PL, 0, 1);
    // z_idt[b,c,d] = sum_n xsq[b,c,n] * bm[b,d,n]   (NT GEMM, K = 16384, direct)
    wmma_gemm_k<float, float><<<grid(RR, IP), blk, 0, stream>>>(
        xsq, (long)IP * NPIX, NPIX, 1, bmb, (long)RR * NPIX, 1, NPIX,
        zid, (long)IP * RR, RR, nullptr, nullptr, nullptr, nullptr, 0, 0,
        IP, RR, NPIX, 0, 0);
    {
        int tot = BSZ * IP * RR;
        adj_z_k<<<dim3((tot + 255) / 256), 256, 0, stream>>>(zid, adj_w, sc_adj, sh_adj, zz);
        wg_z_k<<<dim3((tot + 255) / 256), 256, 0, stream>>>(zz, wg_w, sc_wg, sh_wg, zf);
    }
    // y[b,c,n] = sum_d zf[b,c,d] * bm[b,d,n]
    wmma_gemm_k<float, float><<<grid(NPIX, IP), blk, 0, stream>>>(
        zf, (long)IP * RR, RR, 1, bmb, (long)RR * NPIX, NPIX, 1, ybf, (long)IP * NPIX, NPIX,
        nullptr, nullptr, nullptr, nullptr, 0, 0, IP, NPIX, RR, 0, 1);
    // g_out = relu(feat + bn3(conv3_w @ y)) -> concat channels [256,512) in bf16
    wmma_gemm_k<float, __bf16><<<grid(NPIX, PL), blk, 0, stream>>>(
        conv3_w, 0, IP, 1, ybf, (long)IP * NPIX, NPIX, 1,
        cat + (size_t)PL * NPIX, (long)2 * PL * NPIX, NPIX,
        nullptr, sc_3, sh_3, feat, (long)PL * NPIX, NPIX, PL, NPIX, IP, 1, 1);

    // -------- final fuse: bn_final(final_w @ concat), K = 512, bf16 operand --------
    wmma_gemm_k<__bf16, float><<<grid(NPIX, PL), blk, 0, stream>>>(
        final_w, 0, 2 * PL, 1, cat, (long)2 * PL * NPIX, NPIX, 1,
        outp, (long)PL * NPIX, NPIX,
        nullptr, sc_fin, sh_fin, nullptr, 0, 0, PL, NPIX, 2 * PL, 0, 1);
}
